// DeepNovoPointNetWithLSTM_84164179132430
// MI455X (gfx1250) — matrix operand order repaired and checked by reading, not compiled, not run on hardware
//
#include <hip/hip_runtime.h>
#include <hip/hip_bf16.h>

// ---------------------------------------------------------------------------
// DeepNovo PointNet + LSTM for MI455X (gfx1250, wave32, WMMA bf16 16x16x32)
// B=16 T=32 V=26 ION=8 N=500 U=64 E=512 H=512 CIN=209 (pad 224)
// ---------------------------------------------------------------------------

typedef __attribute__((ext_vector_type(16))) __bf16 v16bf;
typedef __attribute__((ext_vector_type(8)))  float  v8f;

union AF { v16bf v; uint4 u[2]; };

#define NT 4   // 16-col tiles per wave

// ---------------------------------------------------------------------------
// Generic GEMM: out[M,N] = A[M,K](bf16,row-major,lda) * W[N,K]^T (bf16,ldb)
//               (+bias[col]) (+Cin preload) ; optional per-column sum/sumsq.
// K-loop is software-pipelined with two STATICALLY-NAMED register buffer
// sets (no runtime-indexed arrays -> no scratch demotion): fragments for
// kb+32 are in flight while the 4 WMMAs for kb execute.
// ---------------------------------------------------------------------------
template<bool WN, bool ADDC, bool STATS, bool HASB, bool OUTB>
__global__ __launch_bounds__(128)
void gemm_k(const __hip_bfloat16* __restrict__ A, int lda,
            const __hip_bfloat16* __restrict__ W, int ldb,
            const float* __restrict__ bias,
            const float* __restrict__ Cin, int ldcin,
            float* __restrict__ outF, __hip_bfloat16* __restrict__ outB, int ldo,
            float* __restrict__ stats, int M, int N, int K)
{
    const int lane = threadIdx.x & 31;
    const int wave = threadIdx.x >> 5;
    const int half = lane >> 4;
    const int l    = lane & 15;

    int rowBase, colBase;
    if (WN) { rowBase = blockIdx.y * 16;              colBase = (blockIdx.x * 4 + wave) * (NT * 16); }
    else    { rowBase = (blockIdx.y * 4 + wave) * 16; colBase = blockIdx.x * (NT * 16); }
    if (rowBase >= M || colBase >= N) return;

    v8f acc[NT];
    if (ADDC) {
        #pragma unroll
        for (int nt = 0; nt < NT; ++nt) {
            int col = colBase + nt * 16 + l;
            #pragma unroll
            for (int r = 0; r < 8; ++r)
                acc[nt][r] = Cin[(size_t)(rowBase + half * 8 + r) * ldcin + col];
        }
    } else {
        #pragma unroll
        for (int nt = 0; nt < NT; ++nt)
            #pragma unroll
            for (int r = 0; r < 8; ++r) acc[nt][r] = 0.f;
    }

    // A lanes: row = rowBase+l ; half 0 holds K {0..7,16..23}, half 1 {8..15,24..31}
    const __hip_bfloat16* arow = A + (size_t)(rowBase + l) * lda + half * 8;
    const __hip_bfloat16* brow = W + (size_t)(colBase + l) * ldb + half * 8;

    AF a0, a1;
    AF b0[NT], b1[NT];

    auto loadK = [&](int kb, AF& a, AF (&b)[NT]) {
        const uint4* pa = reinterpret_cast<const uint4*>(arow + kb);
        a.u[0] = pa[0];          // 8 bf16
        a.u[1] = pa[2];          // +16 elements
        #pragma unroll
        for (int nt = 0; nt < NT; ++nt) {
            const uint4* pb = reinterpret_cast<const uint4*>(brow + (size_t)(nt * 16) * ldb + kb);
            b[nt].u[0] = pb[0];
            b[nt].u[1] = pb[2];
        }
    };
    auto mma = [&](const AF& a, const AF (&b)[NT]) {
        #pragma unroll
        for (int nt = 0; nt < NT; ++nt)
            acc[nt] = __builtin_amdgcn_wmma_f32_16x16x32_bf16(
                false, a.v, false, b[nt].v, (short)0, acc[nt], false, false);
    };

    loadK(0, a0, b0);
    int kb = 0;
    for (;;) {
        if (kb + 32 < K) loadK(kb + 32, a1, b1);   // prefetch next slice (regs)
        mma(a0, b0);
        kb += 32;
        if (kb >= K) break;
        if (kb + 32 < K) loadK(kb + 32, a0, b0);
        mma(a1, b1);
        kb += 32;
        if (kb >= K) break;
    }

    // Epilogue: C/D layout -> lane half 0: (M=r, N=l), half 1: (M=r+8, N=l)
    #pragma unroll
    for (int nt = 0; nt < NT; ++nt) {
        int col = colBase + nt * 16 + l;
        float bs = HASB ? bias[col] : 0.f;
        float s = 0.f, q = 0.f;
        #pragma unroll
        for (int r = 0; r < 8; ++r) {
            float v = acc[nt][r] + bs;
            size_t idx = (size_t)(rowBase + half * 8 + r) * ldo + col;
            if (OUTB) outB[idx] = __float2bfloat16(v);
            else      outF[idx] = v;
            s += v; q += v * v;
        }
        if (STATS) {
            s += __shfl_xor(s, 16, 32);
            q += __shfl_xor(q, 16, 32);
            if (half == 0) {
                atomicAdd(&stats[col * 2 + 0], s);
                atomicAdd(&stats[col * 2 + 1], q);
            }
        }
    }
}

// ---------------------------------------------------------------------------
// Feature construction + BN0 statistics.  feat: [m*500+n][224] bf16
// ---------------------------------------------------------------------------
__global__ __launch_bounds__(224)
void feat_k(const float* __restrict__ li, const float* __restrict__ pl,
            const float* __restrict__ pi, const float* __restrict__ dsfp,
            __hip_bfloat16* __restrict__ feat, float* __restrict__ stats0)
{
    const int m  = blockIdx.x;        // b*32+t
    const int b  = m >> 5;
    const int c  = threadIdx.x;       // 0..223
    const int n0 = blockIdx.y * 125;
    const float sig = 1.f / (1.f + __expf(-dsfp[0]));
    const float liv = (c < 208) ? li[(size_t)m * 208 + c] : 0.f;
    const float ml  = (liv > 1e-5f) ? 1.f : 0.f;
    float s = 0.f, q = 0.f;
    for (int n = n0; n < n0 + 125; ++n) {
        float v;
        if (c < 208) {
            float plv = pl[b * 500 + n];
            float mp  = (plv > 1e-5f) ? 1.f : 0.f;
            float ppm = (plv - liv) / (liv + 1e-6f) * 1e6f;
            v = __expf(-fabsf(ppm * 0.2f * sig)) * mp * ml;
        } else if (c == 208) {
            v = pi[b * 500 + n];
        } else v = 0.f;
        feat[((size_t)m * 500 + n) * 224 + c] = __float2bfloat16(v);
        s += v; q += v * v;
    }
    if (c < 209) {
        atomicAdd(&stats0[c * 2 + 0], s);
        atomicAdd(&stats0[c * 2 + 1], q);
    }
}

// Fold BN0 into conv1 weights/bias.
__global__ __launch_bounds__(64)
void fold0_k(const float* __restrict__ stats0, const float* __restrict__ w1,
             const float* __restrict__ b1, const float* __restrict__ g0,
             const float* __restrict__ bb0,
             __hip_bfloat16* __restrict__ w1eff, float* __restrict__ b1eff)
{
    const int o = threadIdx.x;
    const float invCnt = 1.f / 256000.f;
    float bacc = b1[o];
    for (int c = 0; c < 224; ++c) {
        if (c < 209) {
            float s = stats0[c * 2], q = stats0[c * 2 + 1];
            float mean = s * invCnt;
            float var  = q * invCnt - mean * mean;
            float sc   = g0[c] * rsqrtf(var + 1e-5f);
            float w    = w1[o * 209 + c];
            w1eff[o * 224 + c] = __float2bfloat16(w * sc);
            bacc += w * (bb0[c] - mean * sc);
        } else {
            w1eff[o * 224 + c] = __float2bfloat16(0.f);
        }
    }
    b1eff[o] = bacc;
}

// In-place BN (batch stats) + ReLU over bf16 activations.
__global__ void bnrelu_k(__hip_bfloat16* __restrict__ buf, const float* __restrict__ stats,
                         const float* __restrict__ g, const float* __restrict__ b,
                         int NC, size_t total, float invCnt)
{
    size_t i = (size_t)blockIdx.x * blockDim.x + threadIdx.x;
    size_t stride = (size_t)gridDim.x * blockDim.x;
    for (; i < total; i += stride) {
        int c = (int)(i % NC);
        float mean = stats[c * 2] * invCnt;
        float var  = stats[c * 2 + 1] * invCnt - mean * mean;
        float sc   = g[c] * rsqrtf(var + 1e-5f);
        float sh   = b[c] - mean * sc;
        float v = __bfloat162float(buf[i]) * sc + sh;
        buf[i] = __float2bfloat16(v > 0.f ? v : 0.f);
    }
}

// Max over n (500) of conv3 activations.
__global__ __launch_bounds__(256)
void pool_k(const __hip_bfloat16* __restrict__ c3, __hip_bfloat16* __restrict__ pooled)
{
    const int m = blockIdx.x, c = threadIdx.x;
    float mx = -3.4e38f;
    for (int n = 0; n < 500; ++n) {
        float v = __bfloat162float(c3[((size_t)m * 500 + n) * 256 + c]);
        mx = v > mx ? v : mx;
    }
    pooled[m * 256 + c] = __float2bfloat16(mx);
}

// Embedding gather, rows ordered t*16+b for the LSTM.
__global__ void emb_k(const int* __restrict__ aa, const float* __restrict__ emb,
                      __hip_bfloat16* __restrict__ xemb)
{
    const int row = blockIdx.x;          // t*16+b
    const int t = row >> 4, b = row & 15;
    const int a = aa[b * 32 + t];
    for (int e = threadIdx.x; e < 512; e += blockDim.x)
        xemb[(size_t)row * 512 + e] = __float2bfloat16(emb[(size_t)a * 512 + e]);
}

__global__ void inithc_k(const float* __restrict__ h0, const float* __restrict__ c0,
                         float* __restrict__ h, float* __restrict__ c,
                         __hip_bfloat16* __restrict__ hbf)
{
    int i = blockIdx.x * blockDim.x + threadIdx.x;   // 8192
    h[i] = h0[i]; c[i] = c0[i];
    hbf[i] = __float2bfloat16(h0[i]);
}

__global__ void cell_k(const float* __restrict__ gates, float* __restrict__ h,
                       float* __restrict__ c, __hip_bfloat16* __restrict__ hbf,
                       float* __restrict__ lstm_relu, int t)
{
    int i = blockIdx.x * blockDim.x + threadIdx.x;   // 0..8191
    int b = i >> 9, hh = i & 511;
    float gi = gates[b * 2048 + hh];
    float gf = gates[b * 2048 + 512 + hh];
    float gg = gates[b * 2048 + 1024 + hh];
    float go = gates[b * 2048 + 1536 + hh];
    float si = 1.f / (1.f + __expf(-gi));
    float sf = 1.f / (1.f + __expf(-gf));
    float so = 1.f / (1.f + __expf(-go));
    float cn = sf * c[i] + si * tanhf(gg);
    float hn = so * tanhf(cn);
    c[i] = cn; h[i] = hn;
    hbf[i] = __float2bfloat16(hn);
    lstm_relu[((size_t)b * 32 + t) * 512 + hh] = hn > 0.f ? hn : 0.f;
}

// logit[r,v] = out_b[v] + ion[r,:64].out_w[v,:64] + relu_h[r,:512].out_w[v,64:]
__global__ __launch_bounds__(32)
void logit_k(const __hip_bfloat16* __restrict__ ion, const float* __restrict__ lstm_relu,
             const float* __restrict__ ow, const float* __restrict__ ob,
             float* __restrict__ out)
{
    const int r = blockIdx.x;    // b*32+t
    const int v = threadIdx.x;
    if (v >= 26) return;
    float acc = ob[v];
    const float* w = ow + v * 576;
    for (int j = 0; j < 64; ++j)
        acc += __bfloat162float(ion[(size_t)r * 64 + j]) * w[j];
    const float* lr = lstm_relu + (size_t)r * 512;
    for (int j = 0; j < 512; ++j) acc += lr[j] * w[64 + j];
    out[r * 26 + v] = acc;
}

__global__ void copyhc_k(const float* __restrict__ h, const float* __restrict__ c,
                         float* __restrict__ out)
{
    int i = blockIdx.x * blockDim.x + threadIdx.x;   // 8192
    out[13312 + i]        = h[i];
    out[13312 + 8192 + i] = c[i];
}

__global__ void zero_k(float* __restrict__ p, int n)
{
    int i = blockIdx.x * blockDim.x + threadIdx.x;
    if (i < n) p[i] = 0.f;
}

__global__ void f2b_k(const float* __restrict__ s, __hip_bfloat16* __restrict__ d, size_t n)
{
    size_t i = (size_t)blockIdx.x * blockDim.x + threadIdx.x;
    size_t st = (size_t)gridDim.x * blockDim.x;
    for (; i < n; i += st) d[i] = __float2bfloat16(s[i]);
}

__global__ void bias2_k(const float* __restrict__ a, const float* __restrict__ b,
                        float* __restrict__ d, int n)
{
    int i = blockIdx.x * blockDim.x + threadIdx.x;
    if (i < n) d[i] = a[i] + b[i];
}

// ---------------------------------------------------------------------------
// Workspace layout (bytes)
// ---------------------------------------------------------------------------
static constexpr size_t OF_FEAT  = 0;                         // 512*500*224*2 = 114,688,000
static constexpr size_t OF_C3    = 0;                         // reuses FEAT (+part of C1), 131,072,000
static constexpr size_t OF_C1    = 114688000;                 // 256000*64*2
static constexpr size_t OF_C2    = 147456000;                 // 256000*128*2
static constexpr size_t OF_STATS = 212992000;                 // 1728 floats
static constexpr size_t OF_W1EFF = OF_STATS + 8192;
static constexpr size_t OF_B1EFF = OF_W1EFF + 28672;
static constexpr size_t OF_W2    = OF_B1EFF + 256;
static constexpr size_t OF_W3    = OF_W2    + 16384;
static constexpr size_t OF_FC1W  = OF_W3    + 65536;
static constexpr size_t OF_FC2W  = OF_FC1W  + 65536;
static constexpr size_t OF_WIH   = OF_FC2W  + 16384;
static constexpr size_t OF_WHH   = OF_WIH   + 2097152;
static constexpr size_t OF_BIASG = OF_WHH   + 2097152;
static constexpr size_t OF_POOL  = OF_BIASG + 8192;
static constexpr size_t OF_FC1O  = OF_POOL  + 262144;
static constexpr size_t OF_ION   = OF_FC1O  + 131072;
static constexpr size_t OF_XEMB  = OF_ION   + 65536;
static constexpr size_t OF_XG    = OF_XEMB  + 524288;
static constexpr size_t OF_GATES = OF_XG    + 4194304;
static constexpr size_t OF_H     = OF_GATES + 131072;
static constexpr size_t OF_C     = OF_H     + 32768;
static constexpr size_t OF_HBF   = OF_C     + 32768;
static constexpr size_t OF_LSTM  = OF_HBF   + 16384;

extern "C" void kernel_launch(void* const* d_in, const int* in_sizes, int n_in,
                              void* d_out, int out_size, void* d_ws, size_t ws_size,
                              hipStream_t stream)
{
    (void)in_sizes; (void)n_in; (void)out_size; (void)ws_size;
    char* w = (char*)d_ws;

    const float* li_in = (const float*)d_in[0];
    const float* pl_in = (const float*)d_in[1];
    const float* pi_in = (const float*)d_in[2];
    const int*   aa_in = (const int*)  d_in[3];
    const float* h0_in = (const float*)d_in[4];
    const float* c0_in = (const float*)d_in[5];
    const float* dsf   = (const float*)d_in[6];
    const float* w1    = (const float*)d_in[7];
    const float* b1    = (const float*)d_in[8];
    const float* w2    = (const float*)d_in[9];
    const float* b2    = (const float*)d_in[10];
    const float* w3    = (const float*)d_in[11];
    const float* b3    = (const float*)d_in[12];
    const float* fc1w  = (const float*)d_in[13];
    const float* fc1b  = (const float*)d_in[14];
    const float* fc2w  = (const float*)d_in[15];
    const float* fc2b  = (const float*)d_in[16];
    const float* g0 = (const float*)d_in[17];  const float* bb0 = (const float*)d_in[18];
    const float* g1 = (const float*)d_in[19];  const float* bb1 = (const float*)d_in[20];
    const float* g2 = (const float*)d_in[21];  const float* bb2 = (const float*)d_in[22];
    const float* g3 = (const float*)d_in[23];  const float* bb3 = (const float*)d_in[24];
    const float* g4 = (const float*)d_in[25];  const float* bb4 = (const float*)d_in[26];
    const float* g5 = (const float*)d_in[27];  const float* bb5 = (const float*)d_in[28];
    const float* emb  = (const float*)d_in[29];
    const float* wih  = (const float*)d_in[30];
    const float* whh  = (const float*)d_in[31];
    const float* bih  = (const float*)d_in[32];
    const float* bhh  = (const float*)d_in[33];
    const float* ow   = (const float*)d_in[34];
    const float* ob   = (const float*)d_in[35];

    __hip_bfloat16* feat  = (__hip_bfloat16*)(w + OF_FEAT);
    __hip_bfloat16* c1b   = (__hip_bfloat16*)(w + OF_C1);
    __hip_bfloat16* c2b   = (__hip_bfloat16*)(w + OF_C2);
    __hip_bfloat16* c3b   = (__hip_bfloat16*)(w + OF_C3);
    float*          stats = (float*)(w + OF_STATS);
    float* s0 = stats;       float* s1 = stats + 448;  float* s2 = stats + 576;
    float* s3 = stats + 832; float* s4 = stats + 1344; float* s5 = stats + 1600;
    __hip_bfloat16* w1eff = (__hip_bfloat16*)(w + OF_W1EFF);
    float*          b1eff = (float*)(w + OF_B1EFF);
    __hip_bfloat16* w2b   = (__hip_bfloat16*)(w + OF_W2);
    __hip_bfloat16* w3b   = (__hip_bfloat16*)(w + OF_W3);
    __hip_bfloat16* fc1wb = (__hip_bfloat16*)(w + OF_FC1W);
    __hip_bfloat16* fc2wb = (__hip_bfloat16*)(w + OF_FC2W);
    __hip_bfloat16* wihb  = (__hip_bfloat16*)(w + OF_WIH);
    __hip_bfloat16* whhb  = (__hip_bfloat16*)(w + OF_WHH);
    float*          biasg = (float*)(w + OF_BIASG);
    __hip_bfloat16* pool  = (__hip_bfloat16*)(w + OF_POOL);
    __hip_bfloat16* fc1o  = (__hip_bfloat16*)(w + OF_FC1O);
    __hip_bfloat16* ion   = (__hip_bfloat16*)(w + OF_ION);
    __hip_bfloat16* xemb  = (__hip_bfloat16*)(w + OF_XEMB);
    float*          Xg    = (float*)(w + OF_XG);
    float*          gates = (float*)(w + OF_GATES);
    float*          hbuf  = (float*)(w + OF_H);
    float*          cbuf  = (float*)(w + OF_C);
    __hip_bfloat16* hbf   = (__hip_bfloat16*)(w + OF_HBF);
    float*          lstmR = (float*)(w + OF_LSTM);
    float*          out   = (float*)d_out;

    // 0. zero BN stats (must be re-zeroed every call)
    zero_k<<<(1728 + 255) / 256, 256, 0, stream>>>(stats, 1728);

    // 1. weight conversions fp32 -> bf16
    f2b_k<<<64,   256, 0, stream>>>(w2,   w2b,   (size_t)128 * 64);
    f2b_k<<<128,  256, 0, stream>>>(w3,   w3b,   (size_t)256 * 128);
    f2b_k<<<128,  256, 0, stream>>>(fc1w, fc1wb, (size_t)128 * 256);
    f2b_k<<<64,   256, 0, stream>>>(fc2w, fc2wb, (size_t)64 * 128);
    f2b_k<<<2048, 256, 0, stream>>>(wih,  wihb,  (size_t)2048 * 512);
    f2b_k<<<2048, 256, 0, stream>>>(whh,  whhb,  (size_t)2048 * 512);
    bias2_k<<<8, 256, 0, stream>>>(bih, bhh, biasg, 2048);

    // 2. features + BN0 stats, then fold BN0 into conv1 weights
    feat_k<<<dim3(512, 4), 224, 0, stream>>>(li_in, pl_in, pi_in, dsf, feat, s0);
    fold0_k<<<1, 64, 0, stream>>>(s0, w1, b1, g0, bb0, w1eff, b1eff);

    // 3. conv1 / conv2 / conv3 (WMMA GEMMs with fused BN-stats epilogue)
    gemm_k<false,false,true,true,true><<<dim3(1, 4000), 128, 0, stream>>>(
        feat, 224, w1eff, 224, b1eff, nullptr, 0, nullptr, c1b, 64, s1, 256000, 64, 224);
    bnrelu_k<<<65536, 256, 0, stream>>>(c1b, s1, g1, bb1, 64, (size_t)256000 * 64, 1.f / 256000.f);

    gemm_k<false,false,true,true,true><<<dim3(2, 4000), 128, 0, stream>>>(
        c1b, 64, w2b, 64, b2, nullptr, 0, nullptr, c2b, 128, s2, 256000, 128, 64);
    bnrelu_k<<<131072, 256, 0, stream>>>(c2b, s2, g2, bb2, 128, (size_t)256000 * 128, 1.f / 256000.f);

    gemm_k<false,false,true,true,true><<<dim3(4, 4000), 128, 0, stream>>>(
        c2b, 128, w3b, 128, b3, nullptr, 0, nullptr, c3b, 256, s3, 256000, 256, 128);
    bnrelu_k<<<262144, 256, 0, stream>>>(c3b, s3, g3, bb3, 256, (size_t)256000 * 256, 1.f / 256000.f);

    // 4. max-pool over N=500, then fc1/fc2 with BN+ReLU
    pool_k<<<512, 256, 0, stream>>>(c3b, pool);

    gemm_k<false,false,true,true,true><<<dim3(2, 8), 128, 0, stream>>>(
        pool, 256, fc1wb, 256, fc1b, nullptr, 0, nullptr, fc1o, 128, s4, 512, 128, 256);
    bnrelu_k<<<256, 256, 0, stream>>>(fc1o, s4, g4, bb4, 128, (size_t)512 * 128, 1.f / 512.f);

    gemm_k<false,false,true,true,true><<<dim3(1, 8), 128, 0, stream>>>(
        fc1o, 128, fc2wb, 128, fc2b, nullptr, 0, nullptr, ion, 64, s5, 512, 64, 128);
    bnrelu_k<<<128, 256, 0, stream>>>(ion, s5, g5, bb5, 64, (size_t)512 * 64, 1.f / 512.f);

    // 5. LSTM: embedding gather, batched input projection Xg, 32 recurrent steps
    emb_k<<<512, 256, 0, stream>>>(aa_in, emb, xemb);
    gemm_k<false,false,false,true,false><<<dim3(32, 8), 128, 0, stream>>>(
        xemb, 512, wihb, 512, biasg, nullptr, 0, Xg, nullptr, 2048, nullptr, 512, 2048, 512);
    inithc_k<<<32, 256, 0, stream>>>(h0_in, c0_in, hbuf, cbuf, hbf);

    for (int t = 0; t < 32; ++t) {
        // gates = h @ Whh^T + Xg[t]   (waves split the 2048 gate columns)
        gemm_k<true,true,false,false,false><<<dim3(8, 1), 128, 0, stream>>>(
            hbf, 512, whhb, 512, nullptr, Xg + (size_t)t * 16 * 2048, 2048,
            gates, nullptr, 2048, nullptr, 16, 2048, 512);
        cell_k<<<32, 256, 0, stream>>>(gates, hbuf, cbuf, hbf, lstmR, t);
    }

    // 6. output projection + (hT, cT)
    logit_k<<<512, 32, 0, stream>>>(ion, lstmR, ow, ob, out);
    copyhc_k<<<32, 256, 0, stream>>>(hbuf, cbuf, out);
}
